// TimmSwinTiny_58617713656396
// MI455X (gfx1250) — matrix operand order repaired and checked by reading, not compiled, not run on hardware
//
#include <hip/hip_runtime.h>
#include <hip/hip_bf16.h>
#include <math.h>

typedef __attribute__((ext_vector_type(16))) _Float16 v16h;
typedef __attribute__((ext_vector_type(8)))  _Float16 v8h;
typedef __attribute__((ext_vector_type(8)))  float    v8f;

#define WMMA(a,b,c) __builtin_amdgcn_wmma_f32_16x16x32_f16(false,(a),false,(b),(short)0,(c),false,false)

__device__ __forceinline__ float gelu_exact(float x) {
  return 0.5f * x * (1.0f + erff(x * 0.70710678118654752440f));
}

// region classification for shifted-window mask (win=7, shift=3):
// slices (0,-7) -> 0, (-7,-3) -> 1, (-3,None) -> 2
__device__ __forceinline__ int regionOf(int r, int n) {
  return (r < n - 7) ? 0 : ((r < n - 3) ? 1 : 2);
}

// ---------------------------------------------------------------------------
// LayerNorm: one wave per row. y = (x - mu) * rsqrt(var + 1e-5) * w + b
// ---------------------------------------------------------------------------
__global__ __launch_bounds__(256) void ln_kernel(
    const float* __restrict__ x, const float* __restrict__ w,
    const float* __restrict__ b, float* __restrict__ y, int M, int C) {
  int row  = blockIdx.x * (blockDim.x >> 5) + (threadIdx.x >> 5);
  int lane = threadIdx.x & 31;
  if (row >= M) return;
  const float* xr = x + (size_t)row * C;
  float s = 0.f, s2 = 0.f;
  for (int c = lane; c < C; c += 32) { float v = xr[c]; s += v; s2 += v * v; }
  #pragma unroll
  for (int o = 16; o; o >>= 1) { s += __shfl_xor(s, o, 32); s2 += __shfl_xor(s2, o, 32); }
  float mu  = s / (float)C;
  float var = s2 / (float)C - mu * mu;
  float inv = rsqrtf(var + 1e-5f);
  float* yr = y + (size_t)row * C;
  for (int c = lane; c < C; c += 32) yr[c] = (xr[c] - mu) * inv * w[c] + b[c];
}

// ---------------------------------------------------------------------------
// WMMA GEMM:  C[m,n] = act( sum_k A[m,k]*W[n,k] + bias[n] ) + resid[m,n]
// A: (M,K) f32 row-major, W: (N,K) f32 row-major.  K % 32 == 0.
// Block = 256 threads (8 waves). Block tile 128x64; wave tile 32x32
// (2 A-frags x 2 B-frags -> 4 WMMA per K-step). Double-buffered LDS with
// register staging hides global-load latency behind the matrix pipe.
// f32 -> f16 conversion at staging; f32 accumulate via WMMA.
// ---------------------------------------------------------------------------
__global__ __launch_bounds__(256) void gemm_kernel(
    const float* __restrict__ A, const float* __restrict__ W,
    const float* __restrict__ bias, const float* resid,
    float* Cmat, int M, int N, int K, int act) {
  __shared__ __align__(16) _Float16 As[2][128 * 40];  // padded stride: conflict-free
  __shared__ __align__(16) _Float16 Bs[2][64 * 40];
  const int tid   = threadIdx.x;
  const int lane  = tid & 31;
  const int wave  = tid >> 5;
  const int wrow  = wave >> 1;          // 0..3 -> 32-row slab
  const int wcol  = wave & 1;           // 0..1 -> 32-col slab
  const int mBase = blockIdx.y * 128;
  const int nBase = blockIdx.x * 64;
  const int lm15  = lane & 15;
  const int khalf = lane >> 4;

  // staging assignments (per thread)
  const int lrowA = tid >> 1;           // 0..127
  const int lcolA = (tid & 1) * 16;     // 0,16
  const int lrowB = tid >> 2;           // 0..63
  const int lcolB = (tid & 3) * 8;      // 0,8,16,24

  const int gmA = mBase + lrowA;
  const int gnB = nBase + lrowB;
  const bool okA = gmA < M;
  const bool okB = gnB < N;
  const float* __restrict__ pA = A + (size_t)(okA ? gmA : (M - 1)) * K;
  const float* __restrict__ pB = W + (size_t)(okB ? gnB : (N - 1)) * K;

  v8f c00 = {}, c01 = {}, c10 = {}, c11 = {};
  float ar[16];
  float br[8];

  const int nk = K >> 5;

  // ---- prologue: fetch + stage tile 0 ----
  {
    #pragma unroll
    for (int t = 0; t < 16; ++t) ar[t] = okA ? pA[lcolA + t] : 0.f;
    #pragma unroll
    for (int t = 0; t < 8; ++t)  br[t] = okB ? pB[lcolB + t] : 0.f;
    _Float16* dA = &As[0][lrowA * 40 + lcolA];
    #pragma unroll
    for (int t = 0; t < 16; ++t) dA[t] = (_Float16)ar[t];
    _Float16* dB = &Bs[0][lrowB * 40 + lcolB];
    #pragma unroll
    for (int t = 0; t < 8; ++t)  dB[t] = (_Float16)br[t];
  }
  __syncthreads();

  for (int s = 0; s < nk; ++s) {
    const int cur = s & 1;
    const int nxt = cur ^ 1;
    const bool more = (s + 1) < nk;

    // ---- prefetch next tile into registers (overlaps with WMMA below) ----
    if (more) {
      const int k1 = (s + 1) << 5;
      #pragma unroll
      for (int t = 0; t < 16; ++t) ar[t] = okA ? pA[k1 + lcolA + t] : 0.f;
      #pragma unroll
      for (int t = 0; t < 8; ++t)  br[t] = okB ? pB[k1 + lcolB + t] : 0.f;
    }

    // ---- fragments from current LDS buffer ----
    v16h af0, af1, bf0, bf1;
    {
      int am0 = 32 * wrow + lm15;
      v8h lo = *(const v8h*)&As[cur][am0 * 40 + khalf * 8];
      v8h hi = *(const v8h*)&As[cur][am0 * 40 + 16 + khalf * 8];
      #pragma unroll
      for (int t = 0; t < 8; ++t) { af0[t] = lo[t]; af0[8 + t] = hi[t]; }
      int am1 = am0 + 16;
      v8h lo1 = *(const v8h*)&As[cur][am1 * 40 + khalf * 8];
      v8h hi1 = *(const v8h*)&As[cur][am1 * 40 + 16 + khalf * 8];
      #pragma unroll
      for (int t = 0; t < 8; ++t) { af1[t] = lo1[t]; af1[8 + t] = hi1[t]; }
      int bn0 = 32 * wcol + lm15;
      v8h ba = *(const v8h*)&Bs[cur][bn0 * 40 + khalf * 8];
      v8h bb = *(const v8h*)&Bs[cur][bn0 * 40 + 16 + khalf * 8];
      #pragma unroll
      for (int t = 0; t < 8; ++t) { bf0[t] = ba[t]; bf0[8 + t] = bb[t]; }
      int bn1 = bn0 + 16;
      v8h ba1 = *(const v8h*)&Bs[cur][bn1 * 40 + khalf * 8];
      v8h bb1 = *(const v8h*)&Bs[cur][bn1 * 40 + 16 + khalf * 8];
      #pragma unroll
      for (int t = 0; t < 8; ++t) { bf1[t] = ba1[t]; bf1[8 + t] = bb1[t]; }
    }

    c00 = WMMA(af0, bf0, c00);
    c01 = WMMA(af0, bf1, c01);
    c10 = WMMA(af1, bf0, c10);
    c11 = WMMA(af1, bf1, c11);

    // ---- stage next tile into alternate LDS buffer ----
    if (more) {
      _Float16* dA = &As[nxt][lrowA * 40 + lcolA];
      #pragma unroll
      for (int t = 0; t < 16; ++t) dA[t] = (_Float16)ar[t];
      _Float16* dB = &Bs[nxt][lrowB * 40 + lcolB];
      #pragma unroll
      for (int t = 0; t < 8; ++t)  dB[t] = (_Float16)br[t];
    }
    __syncthreads();
  }

  // ---- epilogue (C/D layout: lane n = lane&15; VGPR r -> m = r + 8*(lane>>4)) ----
  const int n0 = nBase + 32 * wcol + lm15;
  const int n1 = n0 + 16;
  const int m0 = mBase + 32 * wrow + 8 * khalf;
  #pragma unroll
  for (int r = 0; r < 8; ++r) {
    int ma = m0 + r;
    int mb = ma + 16;
    if (ma < M) {
      if (n0 < N) {
        float v = c00[r] + (bias ? bias[n0] : 0.f);
        if (act == 1) v = gelu_exact(v);
        if (resid) v += resid[(size_t)ma * N + n0];
        Cmat[(size_t)ma * N + n0] = v;
      }
      if (n1 < N) {
        float v = c01[r] + (bias ? bias[n1] : 0.f);
        if (act == 1) v = gelu_exact(v);
        if (resid) v += resid[(size_t)ma * N + n1];
        Cmat[(size_t)ma * N + n1] = v;
      }
    }
    if (mb < M) {
      if (n0 < N) {
        float v = c10[r] + (bias ? bias[n0] : 0.f);
        if (act == 1) v = gelu_exact(v);
        if (resid) v += resid[(size_t)mb * N + n0];
        Cmat[(size_t)mb * N + n0] = v;
      }
      if (n1 < N) {
        float v = c11[r] + (bias ? bias[n1] : 0.f);
        if (act == 1) v = gelu_exact(v);
        if (resid) v += resid[(size_t)mb * N + n1];
        Cmat[(size_t)mb * N + n1] = v;
      }
    }
  }
}

// ---------------------------------------------------------------------------
// Patch embed: 4x4 stride-4 conv as per-token dot product -> (B,56,56,96)
// ---------------------------------------------------------------------------
__global__ void patch_embed_kernel(const float* __restrict__ img,
                                   const float* __restrict__ pw,
                                   const float* __restrict__ pb,
                                   float* __restrict__ y) {
  size_t idx = (size_t)blockIdx.x * blockDim.x + threadIdx.x;
  size_t total = (size_t)32 * 56 * 56 * 96;
  if (idx >= total) return;
  int c = (int)(idx % 96); size_t r = idx / 96;
  int ow = (int)(r % 56); r /= 56;
  int oh = (int)(r % 56); int b = (int)(r / 56);
  const float* ib = img + (size_t)b * 3 * 224 * 224;
  const float* wr = pw + (size_t)c * 48;
  float s = pb[c];
  #pragma unroll
  for (int ci = 0; ci < 3; ++ci)
    #pragma unroll
    for (int kh = 0; kh < 4; ++kh)
      #pragma unroll
      for (int kw = 0; kw < 4; ++kw)
        s += ib[((size_t)ci * 224 + oh * 4 + kh) * 224 + ow * 4 + kw] *
             wr[(ci * 4 + kh) * 4 + kw];
  y[idx] = s;
}

// ---------------------------------------------------------------------------
// Gather qkv (token layout, N=3C) into padded window-head layout, f16.
// Q pre-scaled by 32^-0.5. Rolled coord r maps to source (r+shift) mod H.
// Layout: (b*nwin+wIdx, head) -> 64x32 tiles, rows >=49 zero.
// ---------------------------------------------------------------------------
__global__ void qkv_gather_kernel(const float* __restrict__ qkv,
                                  _Float16* __restrict__ Qh,
                                  _Float16* __restrict__ Kh,
                                  _Float16* __restrict__ Vh,
                                  int B, int H, int W, int Cdim, int heads, int shift) {
  int nw = W / 7, nwin = (H / 7) * nw;
  size_t idx = (size_t)blockIdx.x * blockDim.x + threadIdx.x;
  size_t total = (size_t)B * nwin * heads * 64 * 32;
  if (idx >= total) return;
  int d = (int)(idx & 31); size_t rest = idx >> 5;
  int pos = (int)(rest & 63); rest >>= 6;
  int head = (int)(rest % heads); rest /= heads;
  int wIdx = (int)(rest % nwin); int b = (int)(rest / nwin);
  size_t o = ((((size_t)(b * nwin + wIdx) * heads + head) * 64) + pos) * 32 + d;
  if (pos >= 49) { Qh[o] = (_Float16)0.f; Kh[o] = (_Float16)0.f; Vh[o] = (_Float16)0.f; return; }
  int wh = wIdx / nw, ww = wIdx % nw;
  int rh = wh * 7 + pos / 7, rw = ww * 7 + pos % 7;
  int oh = rh + shift; if (oh >= H) oh -= H;
  int ow = rw + shift; if (ow >= W) ow -= W;
  const float* src = qkv + (((size_t)b * H + oh) * W + ow) * (3 * Cdim) + head * 32 + d;
  Qh[o] = (_Float16)(src[0] * 0.17677669529663687f);  // hd^-0.5, hd = 32
  Kh[o] = (_Float16)src[Cdim];
  Vh[o] = (_Float16)src[2 * Cdim];
}

// ---------------------------------------------------------------------------
// Windowed attention, one 4-wave block per (window, head).
//  S = Q K^T (WMMA, 64x64 padded) -> +rel-pos bias, +shift mask, softmax
//  O = P V   (WMMA) -> scatter directly to token layout via inverse roll.
// ---------------------------------------------------------------------------
__global__ __launch_bounds__(128) void attn_kernel(
    const _Float16* __restrict__ Qh, const _Float16* __restrict__ Kh,
    const _Float16* __restrict__ Vh, const float* __restrict__ rpb,
    float* __restrict__ out, int B, int H, int W, int heads, int shift) {
  __shared__ __align__(16) float Sf[64 * 68];   // padded stride -> conflict-free
  int nw = W / 7, nwin = (H / 7) * nw;
  int gid  = blockIdx.x;            // (b*nwin + wIdx)*heads + head
  int head = gid % heads;
  int wb   = gid / heads;
  int wIdx = wb % nwin;
  int b    = wb / nwin;
  int wh   = wIdx / nw, ww = wIdx % nw;
  const int lane  = threadIdx.x & 31;
  const int wave  = threadIdx.x >> 5;   // 0..3
  const int lm15  = lane & 15;
  const int khalf = lane >> 4;
  size_t base = (size_t)gid * (64 * 32);

  // ---- S = Q K^T ----
  v16h afrag;
  {
    const _Float16* qp = Qh + base + (size_t)(16 * wave + lm15) * 32;
    #pragma unroll
    for (int t = 0; t < 8; ++t) {
      afrag[t]     = qp[khalf * 8 + t];
      afrag[8 + t] = qp[16 + khalf * 8 + t];
    }
  }
  #pragma unroll
  for (int jb = 0; jb < 4; ++jb) {
    const _Float16* kp = Kh + base + (size_t)(jb * 16 + lm15) * 32;
    v16h bfrag;
    #pragma unroll
    for (int t = 0; t < 8; ++t) {
      bfrag[t]     = kp[khalf * 8 + t];
      bfrag[8 + t] = kp[16 + khalf * 8 + t];
    }
    v8f acc = {};
    acc = WMMA(afrag, bfrag, acc);
    int n  = jb * 16 + lm15;
    int mr = 16 * wave + 8 * khalf;
    #pragma unroll
    for (int r = 0; r < 8; ++r) Sf[(mr + r) * 68 + n] = acc[r];
  }
  __syncthreads();

  // ---- bias + mask + softmax (one thread per row) ----
  if (threadIdx.x < 64) {
    int i = threadIdx.x;
    int ih = i / 7, iw = i % 7;
    int regI = 0;
    if (shift) regI = regionOf(wh * 7 + ih, H) * 3 + regionOf(ww * 7 + iw, W);
    float maxv = -1e30f;
    for (int j = 0; j < 64; ++j) {
      float v = Sf[i * 68 + j];
      if (i < 49 && j < 49) {
        int jh = j / 7, jw = j % 7;
        v += rpb[((ih - jh + 6) * 13 + (iw - jw + 6)) * heads + head];
        if (shift) {
          int regJ = regionOf(wh * 7 + jh, H) * 3 + regionOf(ww * 7 + jw, W);
          if (regJ != regI) v -= 100.f;
        }
      }
      if (j >= 49) v = -1e30f;
      Sf[i * 68 + j] = v;
      maxv = fmaxf(maxv, v);
    }
    float s = 0.f;
    for (int j = 0; j < 64; ++j) {
      float e = __expf(Sf[i * 68 + j] - maxv);
      Sf[i * 68 + j] = e;
      s += e;
    }
    float inv = 1.f / s;
    for (int j = 0; j < 64; ++j) Sf[i * 68 + j] *= inv;
  }
  __syncthreads();

  // ---- O = P V, scatter to token layout ----
  const _Float16* vbp = Vh + base;
  int Cdim = heads * 32;
  #pragma unroll
  for (int nb2 = 0; nb2 < 2; ++nb2) {
    v8f acc = {};
    #pragma unroll
    for (int ks = 0; ks < 2; ++ks) {
      v16h pa;
      int m = 16 * wave + lm15;
      #pragma unroll
      for (int t = 0; t < 8; ++t) {
        pa[t]     = (_Float16)Sf[m * 68 + ks * 32 + khalf * 8 + t];
        pa[8 + t] = (_Float16)Sf[m * 68 + ks * 32 + 16 + khalf * 8 + t];
      }
      v16h vb;
      int d = nb2 * 16 + lm15;
      #pragma unroll
      for (int t = 0; t < 8; ++t) {
        vb[t]     = vbp[(size_t)(ks * 32 + khalf * 8 + t) * 32 + d];
        vb[8 + t] = vbp[(size_t)(ks * 32 + 16 + khalf * 8 + t) * 32 + d];
      }
      acc = WMMA(pa, vb, acc);
    }
    int d  = nb2 * 16 + lm15;
    int mr = 16 * wave + 8 * khalf;
    #pragma unroll
    for (int r = 0; r < 8; ++r) {
      int m = mr + r;
      if (m < 49) {
        int rh = wh * 7 + m / 7, rw = ww * 7 + m % 7;
        int oh = rh + shift; if (oh >= H) oh -= H;
        int ow = rw + shift; if (ow >= W) ow -= W;
        out[(((size_t)b * H + oh) * W + ow) * Cdim + head * 32 + d] = acc[r];
      }
    }
  }
}

// ---------------------------------------------------------------------------
// Patch merging gather: (B,H,W,C) -> (B, H/2*W/2, 4C), blocks [00,10,01,11]
// ---------------------------------------------------------------------------
__global__ void merge_gather_kernel(const float* __restrict__ x, float* __restrict__ y,
                                    int B, int H, int W, int Cdim) {
  size_t idx = (size_t)blockIdx.x * blockDim.x + threadIdx.x;
  size_t total = (size_t)B * (H / 2) * (W / 2) * 4 * Cdim;
  if (idx >= total) return;
  int c = (int)(idx % Cdim); size_t r = idx / Cdim;
  int j = (int)(r % 4); r /= 4;
  int ow = (int)(r % (W / 2)); r /= (W / 2);
  int oh = (int)(r % (H / 2)); int b = (int)(r / (H / 2));
  int ih = oh * 2 + (j & 1), iw = ow * 2 + (j >> 1);
  y[idx] = x[(((size_t)b * H + ih) * W + iw) * Cdim + c];
}

// mean + max pooling over tokens
__global__ void pool_kernel(const float* __restrict__ f, float* __restrict__ o,
                            int B, int L, int C) {
  int idx = blockIdx.x * blockDim.x + threadIdx.x;
  if (idx >= B * C) return;
  int c = idx % C, b = idx / C;
  const float* p = f + (size_t)b * L * C + c;
  float s = 0.f, mx = -1e30f;
  for (int l = 0; l < L; ++l) { float v = p[(size_t)l * C]; s += v; mx = fmaxf(mx, v); }
  o[idx] = s / (float)L + mx;
}

// row-wise L2 normalize (in place): x / sqrt(sum(x^2) + 1e-12)
__global__ __launch_bounds__(256) void l2norm_kernel(float* x, int C) {
  __shared__ float red[8];
  __shared__ float invs;
  int b = blockIdx.x;
  float* r = x + (size_t)b * C;
  float s = 0.f;
  for (int c = threadIdx.x; c < C; c += blockDim.x) { float v = r[c]; s += v * v; }
  #pragma unroll
  for (int o = 16; o; o >>= 1) s += __shfl_xor(s, o, 32);
  if ((threadIdx.x & 31) == 0) red[threadIdx.x >> 5] = s;
  __syncthreads();
  if (threadIdx.x == 0) {
    float t = 0.f;
    for (int i = 0; i < 8; ++i) t += red[i];
    invs = 1.f / sqrtf(t + 1e-12f);
  }
  __syncthreads();
  float inv = invs;
  for (int c = threadIdx.x; c < C; c += blockDim.x) r[c] *= inv;
}

// head attention combine: logits = x @ aw^T, softmax(4), weighted sum -> (B,512)
__global__ __launch_bounds__(128) void head_combine_kernel(
    const float* __restrict__ x, const float* __restrict__ aw,
    float* __restrict__ out, int B) {
  __shared__ float lg[4];
  int b = blockIdx.x;
  int wave = threadIdx.x >> 5, lane = threadIdx.x & 31;
  const float* xr = x + (size_t)b * 2048;
  float s = 0.f;
  for (int c = lane; c < 2048; c += 32) s += xr[c] * aw[wave * 2048 + c];
  #pragma unroll
  for (int o = 16; o; o >>= 1) s += __shfl_xor(s, o, 32);
  if (lane == 0) lg[wave] = s;
  __syncthreads();
  float m = fmaxf(fmaxf(lg[0], lg[1]), fmaxf(lg[2], lg[3]));
  float e0 = __expf(lg[0] - m), e1 = __expf(lg[1] - m);
  float e2 = __expf(lg[2] - m), e3 = __expf(lg[3] - m);
  float inv = 1.f / (e0 + e1 + e2 + e3);
  float p0 = e0 * inv, p1 = e1 * inv, p2 = e2 * inv, p3 = e3 * inv;
  for (int e = threadIdx.x; e < 512; e += blockDim.x) {
    out[(size_t)b * 512 + e] =
        xr[e] * p0 + xr[512 + e] * p1 + xr[1024 + e] * p2 + xr[1536 + e] * p3;
  }
}

// ---------------------------------------------------------------------------
// Host orchestration
// ---------------------------------------------------------------------------
struct BlkP {
  const float *n1w, *n1b, *qkvw, *qkvb, *rpb, *projw, *projb;
  const float *n2w, *n2b, *fc1w, *fc1b, *fc2w, *fc2b;
};

static inline void launch_gemm(const float* A, const float* Wt, const float* bias,
                               const float* resid, float* C, int M, int N, int K,
                               int act, hipStream_t s) {
  dim3 g((unsigned)((N + 63) / 64), (unsigned)((M + 127) / 128));
  gemm_kernel<<<g, 256, 0, s>>>(A, Wt, bias, resid, C, M, N, K, act);
}

static inline void launch_ln(const float* x, const float* w, const float* b,
                             float* y, int M, int C, hipStream_t s) {
  ln_kernel<<<(unsigned)((M + 7) / 8), 256, 0, s>>>(x, w, b, y, M, C);
}

extern "C" void kernel_launch(void* const* d_in, const int* in_sizes, int n_in,
                              void* d_out, int out_size, void* d_ws, size_t ws_size,
                              hipStream_t stream) {
  (void)in_sizes; (void)n_in; (void)out_size; (void)ws_size;
  int cur = 0;
  auto nxt = [&]() { return (const float*)d_in[cur++]; };

  const float* x_img   = nxt();
  const float* patch_w = nxt();
  const float* patch_b = nxt();
  const float* pe_nw   = nxt();
  const float* pe_nb   = nxt();

  const int depth[4] = {2, 2, 6, 2};
  const int headsA[4] = {3, 6, 12, 24};
  BlkP blocks[12];
  const float *ds_nw[3], *ds_nb[3], *ds_red[3];
  int bidx = 0;
  for (int si = 0; si < 4; ++si) {
    for (int d = 0; d < depth[si]; ++d) {
      BlkP& p = blocks[bidx++];
      p.n1w = nxt(); p.n1b = nxt();
      p.qkvw = nxt(); p.qkvb = nxt();
      p.rpb = nxt();
      p.projw = nxt(); p.projb = nxt();
      p.n2w = nxt(); p.n2b = nxt();
      p.fc1w = nxt(); p.fc1b = nxt();
      p.fc2w = nxt(); p.fc2b = nxt();
    }
    if (si < 3) { ds_nw[si] = nxt(); ds_nb[si] = nxt(); ds_red[si] = nxt(); }
  }
  const float* fin_nw  = nxt();
  const float* fin_nb  = nxt();
  const float* embed_w = nxt();
  const float* embed_b = nxt();
  const float* attn_w  = nxt();

  // workspace layout
  char* ws = (char*)d_ws;
  size_t off = 0;
  auto carve = [&](size_t bytes) -> char* {
    char* p = ws + off;
    off += (bytes + 255) & ~(size_t)255;
    return p;
  };
  const size_t MTOK = 100352;  // 32 * 56 * 56
  float* X  = (float*)carve(MTOK * 96 * 4);   // current activations (B,H,W,C)
  float* T1 = (float*)carve(MTOK * 96 * 4);   // LN / gather scratch
  float* T2 = (float*)carve(MTOK * 384 * 4);  // qkv out / MLP hidden / merged-LN
  float* T3 = (float*)carve(MTOK * 96 * 4);   // attn token-layout out / embed out
  _Float16* Qh = (_Float16*)carve((size_t)6144 * 64 * 32 * 2);
  _Float16* Kh = (_Float16*)carve((size_t)6144 * 64 * 32 * 2);
  _Float16* Vh = (_Float16*)carve((size_t)6144 * 64 * 32 * 2);

  const int B = 32;

  // patch embed -> T1, then LN -> X
  {
    size_t total = (size_t)B * 56 * 56 * 96;
    patch_embed_kernel<<<(unsigned)((total + 255) / 256), 256, 0, stream>>>(
        x_img, patch_w, patch_b, T1);
    launch_ln(T1, pe_nw, pe_nb, X, (int)(B * 56 * 56), 96, stream);
  }

  int H = 56, W = 56, C = 96;
  bidx = 0;
  for (int si = 0; si < 4; ++si) {
    int heads = headsA[si];
    for (int bi = 0; bi < depth[si]; ++bi) {
      const BlkP& p = blocks[bidx++];
      int shift = (bi % 2 == 1 && H > 7) ? 3 : 0;
      int M = B * H * W;
      int nwin = (H / 7) * (W / 7);

      // LN1 + qkv GEMM
      launch_ln(X, p.n1w, p.n1b, T1, M, C, stream);
      launch_gemm(T1, p.qkvw, p.qkvb, nullptr, T2, M, 3 * C, C, 0, stream);

      // window gather (roll + partition + head split + pad + Q scale)
      {
        size_t total = (size_t)B * nwin * heads * 64 * 32;
        qkv_gather_kernel<<<(unsigned)((total + 255) / 256), 256, 0, stream>>>(
            T2, Qh, Kh, Vh, B, H, W, C, heads, shift);
      }
      // attention (scatters directly to token layout, inverse roll)
      attn_kernel<<<(unsigned)(B * nwin * heads), 128, 0, stream>>>(
          Qh, Kh, Vh, p.rpb, T3, B, H, W, heads, shift);

      // proj + residual
      launch_gemm(T3, p.projw, p.projb, X, X, M, C, C, 0, stream);

      // MLP
      launch_ln(X, p.n2w, p.n2b, T1, M, C, stream);
      launch_gemm(T1, p.fc1w, p.fc1b, nullptr, T2, M, 4 * C, C, 1, stream);
      launch_gemm(T2, p.fc2w, p.fc2b, X, X, M, C, 4 * C, 0, stream);
    }
    if (si < 3) {
      size_t total = (size_t)B * (H / 2) * (W / 2) * 4 * C;
      merge_gather_kernel<<<(unsigned)((total + 255) / 256), 256, 0, stream>>>(
          X, T1, B, H, W, C);
      int M2 = B * (H / 2) * (W / 2);
      launch_ln(T1, ds_nw[si], ds_nb[si], T2, M2, 4 * C, stream);
      launch_gemm(T2, ds_red[si], nullptr, nullptr, X, M2, 2 * C, 4 * C, 0, stream);
      H /= 2; W /= 2; C *= 2;
    }
  }

  // final LN -> T1 (B,49,768)
  launch_ln(X, fin_nw, fin_nb, T1, B * H * W, C, stream);

  // head: mean+max pool -> T2 (B,768)
  pool_kernel<<<(unsigned)((B * 768 + 255) / 256), 256, 0, stream>>>(
      T1, T2, B, 49, 768);
  // embed GEMM -> T3 (B,2048)
  launch_gemm(T2, embed_w, embed_b, nullptr, T3, B, 2048, 768, 0, stream);
  // L2 normalize rows
  l2norm_kernel<<<(unsigned)B, 256, 0, stream>>>(T3, 2048);
  // attention-pool combine -> d_out (B,512)
  head_combine_kernel<<<(unsigned)B, 128, 0, stream>>>(
      T3, attn_w, (float*)d_out, B);
}
